// MultiHeadAttention_16767552323942
// MI455X (gfx1250) — matrix operand order repaired and checked by reading, not compiled, not run on hardware
//
#include <hip/hip_runtime.h>

// ---------- types ----------
typedef __bf16 v16bf __attribute__((ext_vector_type(16)));
typedef float  v8f   __attribute__((ext_vector_type(8)));
typedef float  v4f   __attribute__((ext_vector_type(4)));
typedef unsigned int v4u __attribute__((ext_vector_type(4)));

union Frag { v4u q[2]; v16bf v; };

__device__ __forceinline__ unsigned short f32_to_bf16(float f) {
  unsigned int u = __float_as_uint(f);
  unsigned int r = u + 0x7FFFu + ((u >> 16) & 1u);
  return (unsigned short)(r >> 16);
}

__device__ __forceinline__ float fast_exp2(float x) {
  return __builtin_amdgcn_exp2f(x);  // v_exp_f32 (hardware exp is base-2)
}

// ---------- convert f32 -> bf16 (packed pairs) ----------
__global__ __launch_bounds__(256) void cvt_bf16_kernel(
    const float* __restrict__ in, unsigned int* __restrict__ out, int n2) {
  int i = blockIdx.x * 256 + threadIdx.x;
  if (i < n2) {
    float a = in[2 * i + 0];
    float b = in[2 * i + 1];
    out[i] = (unsigned int)f32_to_bf16(a) | ((unsigned int)f32_to_bf16(b) << 16);
  }
}

// ---------- bf16 GEMM, 128x64 tile, double-buffered LDS ----------
// BF16_OUT=1: store bf16. BF16_OUT=0: store f32 + bias[col].
template <int BF16_OUT>
__global__ __launch_bounds__(128) void gemm_bf16_kernel(
    const unsigned short* __restrict__ A, const unsigned short* __restrict__ B,
    void* __restrict__ C, const float* __restrict__ bias, int M, int N, int K) {
  constexpr int LST = 40;  // lds row stride (halves): 80B, 16B-aligned + bank skew
  __shared__ __align__(16) unsigned short As[2][128 * LST];
  __shared__ __align__(16) unsigned short Bs[2][64 * LST];

  const int tid = threadIdx.x;
  const int wave = tid >> 5, lane = tid & 31;
  const int h = lane >> 4, lm = lane & 15;
  const int m0 = blockIdx.y * 128;
  const int n0 = blockIdx.x * 64;

  v8f acc[2][4] = {};
  v4u areg[4], breg[2];

  auto gload = [&](int k0) {
#pragma unroll
    for (int rep = 0; rep < 4; ++rep) {  // A tile 128x32: 512 8-half chunks
      int c = tid + rep * 128;
      int row = c >> 2;
      int col8 = (c & 3) * 8;
      areg[rep] = *reinterpret_cast<const v4u*>(&A[(size_t)(m0 + row) * K + k0 + col8]);
    }
#pragma unroll
    for (int rep = 0; rep < 2; ++rep) {  // B tile 32x64: 256 chunks
      int c = tid + rep * 128;
      int k = c >> 3;
      int nc = (c & 7) * 8;
      breg[rep] = *reinterpret_cast<const v4u*>(&B[(size_t)(k0 + k) * N + n0 + nc]);
    }
  };

  gload(0);
  for (int k0 = 0; k0 < K; k0 += 32) {
    const int buf = (k0 >> 5) & 1;  // LDS-only dynamic index (fine)
    // stage regs -> LDS
#pragma unroll
    for (int rep = 0; rep < 4; ++rep) {
      int c = tid + rep * 128;
      *reinterpret_cast<v4u*>(&As[buf][(c >> 2) * LST + (c & 3) * 8]) = areg[rep];
    }
#pragma unroll
    for (int rep = 0; rep < 2; ++rep) {
      int c = tid + rep * 128;
      int k = c >> 3;
      int nc = (c & 7) * 8;
      const unsigned short* e = reinterpret_cast<const unsigned short*>(&breg[rep]);
#pragma unroll
      for (int i = 0; i < 8; ++i) Bs[buf][(nc + i) * LST + k] = e[i];  // transpose
    }
    __syncthreads();
    if (k0 + 32 < K) gload(k0 + 32);  // next slab in flight during WMMAs

    Frag af[2];
#pragma unroll
    for (int s = 0; s < 2; ++s) {
      const unsigned short* rowp = &As[buf][(32 * wave + 16 * s + lm) * LST];
      af[s].q[0] = *reinterpret_cast<const v4u*>(rowp + 8 * h);
      af[s].q[1] = *reinterpret_cast<const v4u*>(rowp + 16 + 8 * h);
    }
#pragma unroll
    for (int j = 0; j < 4; ++j) {
      Frag bf;
      const unsigned short* colp = &Bs[buf][(16 * j + lm) * LST];
      bf.q[0] = *reinterpret_cast<const v4u*>(colp + 8 * h);
      bf.q[1] = *reinterpret_cast<const v4u*>(colp + 16 + 8 * h);
#pragma unroll
      for (int s = 0; s < 2; ++s)
        acc[s][j] = __builtin_amdgcn_wmma_f32_16x16x32_bf16(
            false, af[s].v, false, bf.v, (short)0, acc[s][j], false, false);
    }
  }
  // epilogue: C layout lane=16h+n, vgpr r -> (row r+8h, col n)
#pragma unroll
  for (int s = 0; s < 2; ++s) {
#pragma unroll
    for (int j = 0; j < 4; ++j) {
#pragma unroll
      for (int r = 0; r < 8; ++r) {
        size_t row = (size_t)(m0 + 32 * wave + 16 * s + r + 8 * h);
        int col = n0 + 16 * j + lm;
        float v = acc[s][j][r];
        if (BF16_OUT) {
          ((unsigned short*)C)[row * N + col] = f32_to_bf16(v);
        } else {
          ((float*)C)[row * N + col] = v + bias[col];
        }
      }
    }
  }
}

// ---------- repack qkv -> Q/K padded [B,H,S,64], V transposed [B,H,64,S] ----------
__global__ __launch_bounds__(256) void repack_qkv_kernel(
    const unsigned short* __restrict__ qkv,  // [B*S, 2304]
    unsigned short* __restrict__ Qp, unsigned short* __restrict__ Kp,
    unsigned short* __restrict__ Vt) {
  size_t i = (size_t)blockIdx.x * 256 + threadIdx.x;  // over B*H*S*64
  int d = (int)(i & 63);
  size_t t = i >> 6;
  int s = (int)(t & 2047);
  t >>= 11;
  int hh = (int)(t & 15);
  int b = (int)(t >> 4);
  size_t row = ((size_t)b * 2048 + s) * 2304;
  unsigned short q = 0, k = 0;
  if (d < 48) {
    q = qkv[row + 0 * 768 + hh * 48 + d];
    k = qkv[row + 1 * 768 + hh * 48 + d];
    unsigned short v = qkv[row + 2 * 768 + hh * 48 + d];
    Vt[(((size_t)b * 16 + hh) * 64 + d) * 2048 + s] = v;
  }
  size_t qi = (((size_t)b * 16 + hh) * 2048 + s) * 64 + d;
  Qp[qi] = q;  // pad dims 48..63 with zero so QK^T inner K=64 is exact
  Kp[qi] = k;
}

// ---------- flash attention: one wave per 16-row Q tile ----------
__global__ __launch_bounds__(128) void attn_kernel(
    const unsigned short* __restrict__ Qp,  // [B*H, S, 64] bf16, hd padded
    const unsigned short* __restrict__ Kp,  // [B*H, S, 64]
    const unsigned short* __restrict__ Vt,  // [B*H, 64, S] (rows 0..47 used)
    unsigned short* __restrict__ O) {       // [B*S, 768] bf16
  constexpr int S = 2048;
  constexpr int SST = 36;  // f32 score-tile row stride: 144B, 16B-aligned + skew
  __shared__ __align__(16) float Sl[4][16 * SST];
  __shared__ __align__(16) float CL[4][16];

  const int wave = threadIdx.x >> 5, lane = threadIdx.x & 31;
  const int h = lane >> 4, lm = lane & 15;
  const int bid = blockIdx.x;  // B*H*32 = 2048 blocks
  const int bh = bid >> 5;
  const int q0 = ((bid & 31) * 4 + wave) * 16;

  const unsigned short* Qb = Qp + (size_t)bh * S * 64;
  const unsigned short* Kb = Kp + (size_t)bh * S * 64;
  const unsigned short* Vb = Vt + (size_t)bh * 64 * S;

  // Q A-fragments for inner-K chunks 0..31 and 32..63 (loaded once)
  Frag qf[2];
  {
    const unsigned short* rowp = Qb + (size_t)(q0 + lm) * 64;
    qf[0].q[0] = *reinterpret_cast<const v4u*>(rowp + 8 * h);
    qf[0].q[1] = *reinterpret_cast<const v4u*>(rowp + 16 + 8 * h);
    qf[1].q[0] = *reinterpret_cast<const v4u*>(rowp + 32 + 8 * h);
    qf[1].q[1] = *reinterpret_cast<const v4u*>(rowp + 48 + 8 * h);
  }

  // scale folded with log2(e): exp(x*s) == exp2(x*s*log2e); hw exp is base-2
  const float qscale = 0.14433756729740643f * 1.4426950408889634f;
  float mcur = -3.0e38f, lcur = 0.0f;  // stats for row `lm` (both halves agree)
  v8f oacc[3] = {};

  // Two *named* fragment sets so buffer selection is compile-time (no scratch).
  Frag kfa[4], kfb[4];

  auto loadK = [&](Frag (&dst)[4], int k0) {
#pragma unroll
    for (int t = 0; t < 2; ++t) {
      const unsigned short* kp = Kb + (size_t)(k0 + 16 * t + lm) * 64;
      dst[t * 2 + 0].q[0] = *reinterpret_cast<const v4u*>(kp + 8 * h);
      dst[t * 2 + 0].q[1] = *reinterpret_cast<const v4u*>(kp + 16 + 8 * h);
      dst[t * 2 + 1].q[0] = *reinterpret_cast<const v4u*>(kp + 32 + 8 * h);
      dst[t * 2 + 1].q[1] = *reinterpret_cast<const v4u*>(kp + 48 + 8 * h);
    }
  };

  // process one 32-key chunk whose K fragments are already in `cur`
  auto process = [&](const Frag (&cur)[4], int k0) {
    // V fragments for this chunk (in flight during score WMMAs + softmax)
    Frag vf[3];
#pragma unroll
    for (int j = 0; j < 3; ++j) {
      const unsigned short* vp = Vb + (size_t)(16 * j + lm) * S + k0 + 8 * h;
      vf[j].q[0] = *reinterpret_cast<const v4u*>(vp);
      vf[j].q[1] = *reinterpret_cast<const v4u*>(vp + 16);
    }
    // scores: two 16-key tiles, inner K = 64
    v8f sacc[2] = {};
    sacc[0] = __builtin_amdgcn_wmma_f32_16x16x32_bf16(
        false, qf[0].v, false, cur[0].v, (short)0, sacc[0], false, false);
    sacc[0] = __builtin_amdgcn_wmma_f32_16x16x32_bf16(
        false, qf[1].v, false, cur[1].v, (short)0, sacc[0], false, false);
    sacc[1] = __builtin_amdgcn_wmma_f32_16x16x32_bf16(
        false, qf[0].v, false, cur[2].v, (short)0, sacc[1], false, false);
    sacc[1] = __builtin_amdgcn_wmma_f32_16x16x32_bf16(
        false, qf[1].v, false, cur[3].v, (short)0, sacc[1], false, false);

    // scaled scores -> LDS row-major (C layout: lane holds (row r+8h, key lm))
#pragma unroll
    for (int r = 0; r < 8; ++r) {
      Sl[wave][(r + 8 * h) * SST + lm]      = sacc[0][r] * qscale;
      Sl[wave][(r + 8 * h) * SST + 16 + lm] = sacc[1][r] * qscale;
    }
    __builtin_amdgcn_wave_barrier();  // DS in-order within wave

    // read back: row `lm`, keys in exact A-fragment order {8h..8h+7, 16+8h..23+8h}
    float sv[16];
    {
      const float* rowp = &Sl[wave][lm * SST];
      v4f a0 = *reinterpret_cast<const v4f*>(rowp + 8 * h);
      v4f a1 = *reinterpret_cast<const v4f*>(rowp + 8 * h + 4);
      v4f a2 = *reinterpret_cast<const v4f*>(rowp + 16 + 8 * h);
      v4f a3 = *reinterpret_cast<const v4f*>(rowp + 20 + 8 * h);
#pragma unroll
      for (int i = 0; i < 4; ++i) {
        sv[i] = a0[i]; sv[4 + i] = a1[i]; sv[8 + i] = a2[i]; sv[12 + i] = a3[i];
      }
    }
    // online softmax stats for this row: 15 VALU max + one xor-16 shuffle
    float mx = sv[0];
#pragma unroll
    for (int i = 1; i < 16; ++i) mx = fmaxf(mx, sv[i]);
    mx = fmaxf(mx, __shfl_xor(mx, 16, 32));  // other half holds other 16 keys
    float mnew = fmaxf(mcur, mx);
    float corr = fast_exp2(mcur - mnew);
    mcur = mnew;

    // exponentials build the P A-fragment directly in registers
    Frag pf;
    float rs = 0.0f;
#pragma unroll
    for (int i = 0; i < 16; ++i) {
      float p = fast_exp2(sv[i] - mnew);
      rs += p;
      pf.v[i] = (__bf16)p;
    }
    rs += __shfl_xor(rs, 16, 32);
    lcur = lcur * corr + rs;

    // redistribute corr (row-per-lane) to C layout via 16-float LDS bounce
    if (h == 0) CL[wave][lm] = corr;
    __builtin_amdgcn_wave_barrier();
    v4f c0 = *reinterpret_cast<const v4f*>(&CL[wave][8 * h]);
    v4f c1 = *reinterpret_cast<const v4f*>(&CL[wave][8 * h + 4]);
#pragma unroll
    for (int r = 0; r < 4; ++r) {
#pragma unroll
      for (int j = 0; j < 3; ++j) {
        oacc[j][r] *= c0[r];
        oacc[j][r + 4] *= c1[r];
      }
    }
    // O += P @ V (hd=48 -> 3 N-tiles)
#pragma unroll
    for (int j = 0; j < 3; ++j)
      oacc[j] = __builtin_amdgcn_wmma_f32_16x16x32_bf16(
          false, pf.v, false, vf[j].v, (short)0, oacc[j], false, false);
  };

  loadK(kfa, 0);
  for (int k0 = 0; k0 < S; k0 += 64) {  // two 32-key chunks, explicit ping-pong
    loadK(kfb, k0 + 32);                // always in range (loop steps by 64)
    __builtin_prefetch(Kb + (size_t)(k0 + 64 + lm) * 64, 0, 1);
    process(kfa, k0);
    if (k0 + 64 < S) loadK(kfa, k0 + 64);
    process(kfb, k0 + 32);
  }

  // redistribute l, normalize, store to [B,S,768] bf16
  if (h == 0) CL[wave][lm] = lcur;
  __builtin_amdgcn_wave_barrier();
  v4f l0 = *reinterpret_cast<const v4f*>(&CL[wave][8 * h]);
  v4f l1 = *reinterpret_cast<const v4f*>(&CL[wave][8 * h + 4]);
  const int b = bh >> 4, hh = bh & 15;
#pragma unroll
  for (int r = 0; r < 8; ++r) {
    float inv = 1.0f / (r < 4 ? l0[r & 3] : l1[r & 3]);
    size_t row = (size_t)b * 2048 + q0 + r + 8 * h;
#pragma unroll
    for (int j = 0; j < 3; ++j)
      O[row * 768 + hh * 48 + 16 * j + lm] = f32_to_bf16(oacc[j][r] * inv);
  }
}

// ---------- host launch ----------
extern "C" void kernel_launch(void* const* d_in, const int* in_sizes, int n_in,
                              void* d_out, int out_size, void* d_ws, size_t ws_size,
                              hipStream_t stream) {
  (void)in_sizes; (void)n_in; (void)out_size; (void)ws_size;
  const float* x      = (const float*)d_in[0];  // [4,2048,768]
  const float* w_qkv  = (const float*)d_in[1];  // [768,2304]
  const float* w_proj = (const float*)d_in[2];  // [768,768]
  const float* b_proj = (const float*)d_in[3];  // [768]

  const int M = 4 * 2048;  // 8192 token rows
  const int D = 768, N_QKV = 2304;

  char* ws = (char*)d_ws;
  unsigned short* xb     = (unsigned short*)(ws);
  unsigned short* wqkvb  = (unsigned short*)(ws + 12582912);
  unsigned short* wprojb = (unsigned short*)(ws + 16121856);
  unsigned short* qkvb   = (unsigned short*)(ws + 17301504);
  unsigned short* Qp     = (unsigned short*)(ws + 55050240);
  unsigned short* Kp     = (unsigned short*)(ws + 71827456);
  unsigned short* Vt     = (unsigned short*)(ws + 88604672);
  unsigned short* attnO  = qkvb;  // reuse: qkv dead after repack

  {
    int n2 = M * D / 2;
    cvt_bf16_kernel<<<(n2 + 255) / 256, 256, 0, stream>>>(x, (unsigned int*)xb, n2);
    n2 = D * N_QKV / 2;
    cvt_bf16_kernel<<<(n2 + 255) / 256, 256, 0, stream>>>(w_qkv, (unsigned int*)wqkvb, n2);
    n2 = D * D / 2;
    cvt_bf16_kernel<<<(n2 + 255) / 256, 256, 0, stream>>>(w_proj, (unsigned int*)wprojb, n2);
  }
  {
    dim3 grid(N_QKV / 64, M / 128);
    gemm_bf16_kernel<1><<<grid, 128, 0, stream>>>(xb, wqkvb, qkvb, nullptr, M, N_QKV, D);
  }
  {
    size_t total = (size_t)4 * 16 * 2048 * 64;
    repack_qkv_kernel<<<(unsigned)(total / 256), 256, 0, stream>>>(qkvb, Qp, Kp, Vt);
  }
  attn_kernel<<<2048, 128, 0, stream>>>(Qp, Kp, Vt, attnO);
  {
    dim3 grid(D / 64, M / 128);
    gemm_bf16_kernel<0><<<grid, 128, 0, stream>>>(attnO, wprojb, d_out, b_proj, M, D, D);
  }
}
